// FastCausalAttention_87479893885257
// MI455X (gfx1250) — matrix operand order repaired and checked by reading, not compiled, not run on hardware
//
#include <hip/hip_runtime.h>
#include <hip/hip_bf16.h>

// ---------------------------------------------------------------------------
// Fused causal attention for MI455X (gfx1250, wave32, WMMA).
// All heavy math in v_wmma_f32_16x16x32_bf16 (fp32 accumulate); softmax/RoPE
// in fp32 VALU. Entire intermediate set (~75MB bf16) fits in 192MB L2, so the
// problem is matrix-pipe bound: bf16 WMMA (K=32/issue) over fp32 WMMA (K=4).
// ---------------------------------------------------------------------------

typedef __bf16 bf16_t;
typedef __attribute__((ext_vector_type(16))) __bf16 v16bf;
typedef __attribute__((ext_vector_type(8)))  __bf16 v8bf;
typedef __attribute__((ext_vector_type(8)))  float  v8f;

union AB16 { v16bf v; v8bf h[2]; };

#define WMMA_BF16(a, b, c) \
  __builtin_amdgcn_wmma_f32_16x16x32_bf16(false, (a), false, (b), (short)0, (c), false, false)

__device__ __forceinline__ v16bf ld16g(const bf16_t* p) { return *(const v16bf*)p; }
__device__ __forceinline__ v8bf  ld8(const bf16_t* p)   { return *(const v8bf*)p; }
__device__ __forceinline__ void  st8(bf16_t* p, v8bf v) { *(v8bf*)p = v; }

static constexpr int Bn  = 2;
static constexpr int Tn  = 2048;
static constexpr int Dn  = 1024;
static constexpr int Hn  = 16;
static constexpr int DHn = 64;

// ---------------------------------------------------------------------------
// fp32 -> bf16 convert (4 elements / thread; all sizes are multiples of 4)
// ---------------------------------------------------------------------------
__global__ void cvt_f32_to_bf16(const float* __restrict__ src,
                                bf16_t* __restrict__ dst, int n)
{
  int i = (blockIdx.x * blockDim.x + threadIdx.x) * 4;
  if (i < n) {
    float4 f = *(const float4*)(src + i);
    dst[i + 0] = (bf16_t)f.x;
    dst[i + 1] = (bf16_t)f.y;
    dst[i + 2] = (bf16_t)f.z;
    dst[i + 3] = (bf16_t)f.w;
  }
}

// ---------------------------------------------------------------------------
// C[M,N] = A[M,K] * B[N,K]^T   (bf16 in, fp32 accumulate, TOut out)
// Block: 256 threads = 8 waves, tile 128(M) x 128(N), K-step 32.
// Wave grid 2x4: each wave owns 64x32 => 4 A-frags x 2 B-frags x 8 wmma/step.
// Double-buffered LDS: next tile's global loads overlap current tile's WMMAs;
// one barrier per K-step. M%128==0, N%128==0, K%32==0 (no bounds checks).
// ---------------------------------------------------------------------------
template <typename TOut>
__global__ __launch_bounds__(256)
void gemm_bf16_tn(const bf16_t* __restrict__ A, const bf16_t* __restrict__ B,
                  TOut* __restrict__ C, int M, int N, int K)
{
  constexpr int LDSD = 40;               // row stride (bf16): 80B, bank-spread
  __shared__ bf16_t As[2][128 * LDSD];
  __shared__ bf16_t Bs[2][128 * LDSD];

  const int tid  = threadIdx.x;
  const int wave = tid >> 5;
  const int lane = tid & 31;
  const int lr   = lane & 15;
  const int half = lane >> 4;            // 0: lanes 0-15, 1: lanes 16-31
  const int m0   = blockIdx.y * 128;
  const int n0   = blockIdx.x * 128;
  const int wm   = (wave >> 2) * 64;     // 0 / 64
  const int wn   = (wave & 3) * 32;      // 0 / 32 / 64 / 96

  v8f acc[4][2];
  for (int i = 0; i < 4; ++i)
    for (int j = 0; j < 2; ++j)
      for (int r = 0; r < 8; ++r) acc[i][j][r] = 0.0f;

  const int ldr = tid >> 1;              // 0..127
  const int ldc = (tid & 1) * 16;        // 0 / 16
  const bf16_t* aptr = A + (size_t)(m0 + ldr) * K + ldc;
  const bf16_t* bptr = B + (size_t)(n0 + ldr) * K + ldc;

  // Prologue: stage tile 0 into buffer 0
  AB16 ra, rb;
  ra.v = ld16g(aptr);
  rb.v = ld16g(bptr);
  st8(&As[0][ldr * LDSD + ldc],     ra.h[0]);
  st8(&As[0][ldr * LDSD + ldc + 8], ra.h[1]);
  st8(&Bs[0][ldr * LDSD + ldc],     rb.h[0]);
  st8(&Bs[0][ldr * LDSD + ldc + 8], rb.h[1]);
  __syncthreads();

  const int nk = K >> 5;
  for (int kt = 0; kt < nk; ++kt) {
    const int cur = kt & 1;
    const int nxt = cur ^ 1;
    const bool more = (kt + 1) < nk;

    // Kick off next tile's global loads (overlap with this tile's WMMAs)
    if (more) {
      ra.v = ld16g(aptr + (kt + 1) * 32);
      rb.v = ld16g(bptr + (kt + 1) * 32);
    }

    // Fragment loads from current buffer (batched, then 8 back-to-back WMMAs)
    v16bf af[4], bfrag[2];
    const int kb = half ? 8 : 0;
    for (int i = 0; i < 4; ++i) {
      const int row = wm + i * 16 + lr;
      AB16 t;
      t.h[0] = ld8(&As[cur][row * LDSD + kb]);
      t.h[1] = ld8(&As[cur][row * LDSD + kb + 16]);
      af[i] = t.v;
    }
    for (int j = 0; j < 2; ++j) {
      const int nr = wn + j * 16 + lr;
      AB16 t;
      t.h[0] = ld8(&Bs[cur][nr * LDSD + half * 8]);
      t.h[1] = ld8(&Bs[cur][nr * LDSD + half * 8 + 16]);
      bfrag[j] = t.v;
    }
    for (int i = 0; i < 4; ++i)
      for (int j = 0; j < 2; ++j)
        acc[i][j] = WMMA_BF16(af[i], bfrag[j], acc[i][j]);

    // Stage next tile into the other buffer (no race: distinct buffer)
    if (more) {
      st8(&As[nxt][ldr * LDSD + ldc],     ra.h[0]);
      st8(&As[nxt][ldr * LDSD + ldc + 8], ra.h[1]);
      st8(&Bs[nxt][ldr * LDSD + ldc],     rb.h[0]);
      st8(&Bs[nxt][ldr * LDSD + ldc + 8], rb.h[1]);
      __syncthreads();
    }
  }

  // C/D layout: VGPR r -> row r + 8*half ; lane lr -> column
  for (int i = 0; i < 4; ++i) {
    const int rbase = m0 + wm + i * 16 + half * 8;
    for (int j = 0; j < 2; ++j) {
      const int col = n0 + wn + j * 16 + lr;
      for (int r = 0; r < 8; ++r)
        C[(size_t)(rbase + r) * N + col] = (TOut)acc[i][j][r];
    }
  }
}

// ---------------------------------------------------------------------------
// RoPE + head split: qkv[B,T,3*H*DH] -> q,k [B,H,T,DH] (rotated), vT [B,H,DH,T]
// One thread per (b,h,t,i<32); handles the (i, i+32) rotation pair.
// ---------------------------------------------------------------------------
__global__ void rope_split(const bf16_t* __restrict__ qkv,
                           bf16_t* __restrict__ q, bf16_t* __restrict__ k,
                           bf16_t* __restrict__ vT)
{
  const int idx = blockIdx.x * blockDim.x + threadIdx.x;   // B*H*T*32
  const int i  = idx & 31;
  const int t  = (idx >> 5) & (Tn - 1);
  const int bh = idx / (Tn * 32);
  const int b  = bh / Hn;
  const int h  = bh % Hn;

  const bf16_t* row = qkv + (size_t)(b * Tn + t) * (3 * Hn * DHn);

  // inv_freq = BASE^(-i/32); angle = t * inv_freq; dims i and i+32 share it
  const float inv = powf(10000.0f, -(float)i * (1.0f / 32.0f));
  const float ang = (float)t * inv;
  const float c = cosf(ang), s = sinf(ang);

  const float q1 = (float)row[h * DHn + i];
  const float q2 = (float)row[h * DHn + 32 + i];
  const float k1 = (float)row[Hn * DHn + h * DHn + i];
  const float k2 = (float)row[Hn * DHn + h * DHn + 32 + i];

  bf16_t* qd = q + ((size_t)bh * Tn + t) * DHn;
  bf16_t* kd = k + ((size_t)bh * Tn + t) * DHn;
  qd[i]      = (bf16_t)(q1 * c - q2 * s);
  qd[i + 32] = (bf16_t)(q2 * c + q1 * s);
  kd[i]      = (bf16_t)(k1 * c - k2 * s);
  kd[i + 32] = (bf16_t)(k2 * c + k1 * s);

  vT[((size_t)bh * DHn + i) * Tn + t]      = row[2 * Hn * DHn + h * DHn + i];
  vT[((size_t)bh * DHn + 32 + i) * Tn + t] = row[2 * Hn * DHn + h * DHn + 32 + i];
}

// ---------------------------------------------------------------------------
// Flash attention (causal). Block = 128 threads = 4 waves, 64-query tile,
// each wave owns 16 query rows. Key tiles of 64 staged in LDS; online softmax
// in fp32 on the WMMA C-fragment layout; P converted C->A layout via LDS.
// Fragment loads are batched per K-step so WMMAs issue back-to-back after a
// single dscnt wait. attn out: [B, T, H*DH] bf16 (scale + normalize applied).
// ---------------------------------------------------------------------------
__global__ __launch_bounds__(128)
void flash_attn_causal(const bf16_t* __restrict__ q, const bf16_t* __restrict__ k,
                       const bf16_t* __restrict__ vT, bf16_t* __restrict__ attn)
{
  constexpr int LDK = 72;                  // tile row stride (bf16), 144B
  __shared__ bf16_t Ks[64 * LDK];          // [key][dh]
  __shared__ bf16_t Vs[64 * LDK];          // [dh][key]   (from vT)
  __shared__ bf16_t Ps[4 * 16 * LDK];      // per-wave P staging [row][key]

  const int tid  = threadIdx.x;
  const int wave = tid >> 5;
  const int lane = tid & 31;
  const int lr   = lane & 15;
  const int half = lane >> 4;

  const int q0 = blockIdx.x * 64;
  const int bh = blockIdx.y;               // b*H + h
  const int b  = bh / Hn;
  const int h  = bh % Hn;

  const bf16_t* qbase = q  + (size_t)bh * Tn * DHn;
  const bf16_t* kbase = k  + (size_t)bh * Tn * DHn;
  const bf16_t* vbase = vT + (size_t)bh * DHn * Tn;

  const int qw = q0 + wave * 16;           // this wave's first query row

  // Load the wave's Q fragments once (A layout, 2 K-steps over DH=64)
  v16bf aq[2];
  {
    const int row = qw + lr;
    const int kb  = half ? 8 : 0;
    for (int kk = 0; kk < 2; ++kk) {
      AB16 t;
      t.h[0] = ld8(qbase + (size_t)row * DHn + kk * 32 + kb);
      t.h[1] = ld8(qbase + (size_t)row * DHn + kk * 32 + kb + 16);
      aq[kk] = t.v;
    }
  }

  v8f   oacc[4];
  float mrow[8], lrow[8];
  for (int nf = 0; nf < 4; ++nf)
    for (int r = 0; r < 8; ++r) oacc[nf][r] = 0.0f;
  for (int r = 0; r < 8; ++r) { mrow[r] = -3.0e38f; lrow[r] = 0.0f; }

  const float scale = 0.125f;              // 1/sqrt(64)
  const int   nkt   = q0 / 64 + 1;         // causal: keys up to q0+63

  for (int kt = 0; kt < nkt; ++kt) {
    const int key0 = kt * 64;

    __syncthreads();                       // protect K/V/P tiles from prior reads
    {
      // 128 threads stage two 64x64 bf16 tiles: 2 threads/row, 32 bf16 each
      const int row = tid >> 1;
      const int col = (tid & 1) * 32;
      AB16 tk0, tk1, tv0, tv1;
      tk0.v = ld16g(kbase + (size_t)(key0 + row) * DHn + col);
      tk1.v = ld16g(kbase + (size_t)(key0 + row) * DHn + col + 16);
      tv0.v = ld16g(vbase + (size_t)row * Tn + key0 + col);
      tv1.v = ld16g(vbase + (size_t)row * Tn + key0 + col + 16);
      st8(&Ks[row * LDK + col],      tk0.h[0]);
      st8(&Ks[row * LDK + col + 8],  tk0.h[1]);
      st8(&Ks[row * LDK + col + 16], tk1.h[0]);
      st8(&Ks[row * LDK + col + 24], tk1.h[1]);
      st8(&Vs[row * LDK + col],      tv0.h[0]);
      st8(&Vs[row * LDK + col + 8],  tv0.h[1]);
      st8(&Vs[row * LDK + col + 16], tv1.h[0]);
      st8(&Vs[row * LDK + col + 24], tv1.h[1]);
    }
    __syncthreads();

    // S = Q * K^T : per K-step load all 4 B-frags, then 4 WMMAs back-to-back
    v8f sacc[4];
    for (int nf = 0; nf < 4; ++nf)
      for (int r = 0; r < 8; ++r) sacc[nf][r] = 0.0f;
    for (int kk = 0; kk < 2; ++kk) {
      v16bf bk[4];
      for (int nf = 0; nf < 4; ++nf) {
        const int nr = nf * 16 + lr;
        AB16 t;
        t.h[0] = ld8(&Ks[nr * LDK + kk * 32 + half * 8]);
        t.h[1] = ld8(&Ks[nr * LDK + kk * 32 + half * 8 + 16]);
        bk[nf] = t.v;
      }
      for (int nf = 0; nf < 4; ++nf)
        sacc[nf] = WMMA_BF16(aq[kk], bk[nf], sacc[nf]);
    }

    // Online softmax in the C-fragment layout:
    //   element (r, nf): row = qw + r + 8*half, key = key0 + nf*16 + lr
    float st[4][8];
    float mnew[8];
    for (int r = 0; r < 8; ++r) mnew[r] = mrow[r];
    for (int nf = 0; nf < 4; ++nf) {
      const int key = key0 + nf * 16 + lr;
      for (int r = 0; r < 8; ++r) {
        const int row = qw + r + half * 8;
        float sv = sacc[nf][r] * scale;
        if (key > row) sv = -3.0e38f;      // causal mask
        st[nf][r] = sv;
        mnew[r] = fmaxf(mnew[r], sv);
      }
    }
    for (int r = 0; r < 8; ++r)            // row-max across the 16-lane half
      for (int off = 1; off < 16; off <<= 1)
        mnew[r] = fmaxf(mnew[r], __shfl_xor(mnew[r], off, 32));

    float alpha[8], psum[8];
    for (int r = 0; r < 8; ++r) {
      alpha[r] = __expf(mrow[r] - mnew[r]);
      mrow[r]  = mnew[r];
      psum[r]  = 0.0f;
    }
    bf16_t* pw = &Ps[wave * 16 * LDK];
    for (int nf = 0; nf < 4; ++nf) {
      const int kloc = nf * 16 + lr;
      for (int r = 0; r < 8; ++r) {
        const float p = __expf(st[nf][r] - mnew[r]);
        psum[r] += p;
        pw[(r + half * 8) * LDK + kloc] = (bf16_t)p;   // C-layout -> LDS
      }
    }
    for (int r = 0; r < 8; ++r) {
      for (int off = 1; off < 16; off <<= 1)
        psum[r] += __shfl_xor(psum[r], off, 32);
      lrow[r] = lrow[r] * alpha[r] + psum[r];
    }
    for (int nf = 0; nf < 4; ++nf)
      for (int r = 0; r < 8; ++r) oacc[nf][r] *= alpha[r];

    __syncthreads();                       // P visible (cross-lane via LDS)

    // O += P * V : batched frag loads per K-step, then 4 WMMAs back-to-back
    const int kb = half ? 8 : 0;
    for (int kk = 0; kk < 2; ++kk) {
      AB16 ap;
      ap.h[0] = ld8(&pw[lr * LDK + kk * 32 + kb]);
      ap.h[1] = ld8(&pw[lr * LDK + kk * 32 + kb + 16]);
      v16bf bv[4];
      for (int nf = 0; nf < 4; ++nf) {
        const int nr = nf * 16 + lr;       // dh row of Vs
        AB16 t;
        t.h[0] = ld8(&Vs[nr * LDK + kk * 32 + half * 8]);
        t.h[1] = ld8(&Vs[nr * LDK + kk * 32 + half * 8 + 16]);
        bv[nf] = t.v;
      }
      for (int nf = 0; nf < 4; ++nf)
        oacc[nf] = WMMA_BF16(ap.v, bv[nf], oacc[nf]);
    }
  }

  // Normalize and write out: attn[b, t, h*64 + dh]
  for (int nf = 0; nf < 4; ++nf) {
    const int col = h * DHn + nf * 16 + lr;
    for (int r = 0; r < 8; ++r) {
      const int row = qw + r + half * 8;
      attn[(size_t)(b * Tn + row) * (Hn * DHn) + col] =
          (bf16_t)(oacc[nf][r] / lrow[r]);
    }
  }
}

// ---------------------------------------------------------------------------
// Launch
// ---------------------------------------------------------------------------
extern "C" void kernel_launch(void* const* d_in, const int* in_sizes, int n_in,
                              void* d_out, int out_size, void* d_ws, size_t ws_size,
                              hipStream_t stream)
{
  const float* x     = (const float*)d_in[0];   // [B,T,D]
  const float* qkv_w = (const float*)d_in[1];   // [3*H*DH, D]
  const float* out_w = (const float*)d_in[2];   // [D, H*DH]
  float* out = (float*)d_out;                   // [B,T,D]

  const size_t MT = (size_t)Bn * Tn;            // 4096 rows
  char*  ws  = (char*)d_ws;
  size_t off = 0;
  auto alloc = [&](size_t bytes) {
    void* p = ws + off;
    off += (bytes + 255) & ~(size_t)255;
    return p;
  };

  bf16_t* x_bf    = (bf16_t*)alloc(MT * Dn * 2);                 // 8 MB
  bf16_t* wq_bf   = (bf16_t*)alloc((size_t)3 * Dn * Dn * 2);     // 6 MB
  bf16_t* wo_bf   = (bf16_t*)alloc((size_t)Dn * Dn * 2);         // 2 MB
  bf16_t* qkv_bf  = (bf16_t*)alloc(MT * 3 * Dn * 2);             // 24 MB
  bf16_t* q_bf    = (bf16_t*)alloc(MT * Dn * 2);                 // 8 MB
  bf16_t* k_bf    = (bf16_t*)alloc(MT * Dn * 2);                 // 8 MB
  bf16_t* vT_bf   = (bf16_t*)alloc(MT * Dn * 2);                 // 8 MB
  bf16_t* attn_bf = (bf16_t*)alloc(MT * Dn * 2);                 // 8 MB
  (void)ws_size; (void)n_in; (void)in_sizes; (void)out_size;

  // 1) fp32 -> bf16
  {
    int nx = (int)(MT * Dn), nw = 3 * Dn * Dn, no = Dn * Dn;
    cvt_f32_to_bf16<<<(nx / 4 + 255) / 256, 256, 0, stream>>>(x, x_bf, nx);
    cvt_f32_to_bf16<<<(nw / 4 + 255) / 256, 256, 0, stream>>>(qkv_w, wq_bf, nw);
    cvt_f32_to_bf16<<<(no / 4 + 255) / 256, 256, 0, stream>>>(out_w, wo_bf, no);
  }

  // 2) QKV projection: [4096,1024] x [3072,1024]^T -> bf16 [4096,3072]
  gemm_bf16_tn<bf16_t><<<dim3(3 * Dn / 128, (int)(MT / 128)), 256, 0, stream>>>(
      x_bf, wq_bf, qkv_bf, (int)MT, 3 * Dn, Dn);

  // 3) RoPE + split into q,k [B,H,T,DH] and vT [B,H,DH,T]
  rope_split<<<(Bn * Hn * Tn * 32) / 256, 256, 0, stream>>>(qkv_bf, q_bf, k_bf, vT_bf);

  // 4) Causal flash attention -> attn_bf [B,T,H*DH]
  flash_attn_causal<<<dim3(Tn / 64, Bn * Hn), 128, 0, stream>>>(q_bf, k_bf, vT_bf, attn_bf);

  // 5) Output projection: [4096,1024] x [1024,1024]^T -> fp32 d_out
  gemm_bf16_tn<float><<<dim3(Dn / 128, (int)(MT / 128)), 256, 0, stream>>>(
      attn_bf, wo_bf, out, (int)MT, Dn, Dn);
}
